// Transformer_20650202759875
// MI455X (gfx1250) — compile-verified, matrix operand order checked by reading, not run on hardware
//
#include <hip/hip_runtime.h>
#include <hip/hip_bf16.h>

// ---------------------------------------------------------------------------
// Graph-transformer forward for MI455X (gfx1250, wave32, WMMA).
// GEMMs in f16 via v_wmma_f32_16x16x32_f16 (fp32 accumulate), double-buffered
// LDS pipeline fed by CDNA5 async global->LDS copies, weights pre-transposed
// so all LDS traffic is b128.
// Attention is wave-per-(token,head) with shuffle softmax over K=32 neighbors.
// ---------------------------------------------------------------------------

typedef __attribute__((ext_vector_type(16))) _Float16 v16h;
typedef __attribute__((ext_vector_type(8)))  _Float16 v8h;
typedef __attribute__((ext_vector_type(8)))  float    v8f;
typedef __attribute__((ext_vector_type(4)))  int      v4i;

#define R_TOK   8192      // B*N
#define FDIM    256
#define DMODEL  256
#define HHEADS  8
#define HDIM    32
#define KNBR    32
#define HIDDEN  512
#define NQKV    768

// Optional CDNA5 async global->LDS path (ASYNCcnt-tracked), guarded so the
// file still compiles on toolchains without the builtin.
#if defined(__gfx1250__) && __has_builtin(__builtin_amdgcn_global_load_async_to_lds_b128) && \
    __has_builtin(__builtin_amdgcn_s_wait_asynccnt)
#define GT_ASYNC 1
#endif

static __device__ __forceinline__ float wred_sum(float v) {
#pragma unroll
  for (int o = 16; o > 0; o >>= 1) v += __shfl_xor(v, o);
  return v;
}

// Load one 16x32 f16 fragment (A layout per CDNA5 ISA 7.12.2) from an LDS row.
// rowp points at element K=0 of that matrix row; hi = lane>>4.
// lane<16 : halves = row[0..7]  ++ row[16..23]
// lane>=16: halves = row[8..15] ++ row[24..31]
static __device__ __forceinline__ v16h frag16(const _Float16* rowp, int hi) {
  v8h a = *(const v8h*)(rowp + hi * 8);
  v8h b = *(const v8h*)(rowp + 16 + hi * 8);
  v16h r;
#pragma unroll
  for (int i = 0; i < 8; ++i) { r[i] = a[i]; r[8 + i] = b[i]; }
  return r;
}

#ifdef GT_ASYNC
// Builtin prototype (from clang diagnostic): (v4i AS1* src, v4i AS3* dst, Ii, Ii)
typedef __attribute__((address_space(1))) v4i gas_v4i;
typedef __attribute__((address_space(3))) v4i las_v4i;
// 16 halves (32B) global -> LDS, two async b128 transfers.
static __device__ __forceinline__ void async_cp16(const _Float16* g, _Float16* l) {
  __builtin_amdgcn_global_load_async_to_lds_b128((gas_v4i*)g,       (las_v4i*)l,       0, 0);
  __builtin_amdgcn_global_load_async_to_lds_b128((gas_v4i*)(g + 8), (las_v4i*)(l + 8), 0, 0);
}
#endif

// ---------------------------------------------------------------------------
// Weight conversion with transpose: src (rows x cols, f32 row-major) ->
// dst (cols x rows, f16 row-major with leading dim dstld). Lets the GEMM stage
// B tiles as contiguous b128 LDS stores (no in-LDS transpose).
// ---------------------------------------------------------------------------
__global__ void gt_cvt_t(const float* __restrict__ src, _Float16* __restrict__ dst,
                         int rows, int cols, int dstld) {
  int i = blockIdx.x * 256 + threadIdx.x;
  if (i >= rows * cols) return;
  int r = i / cols, c = i - r * cols;
  dst[(size_t)c * dstld + r] = (_Float16)src[i];
}

__global__ void gt_bias3(const float* __restrict__ a, const float* __restrict__ b,
                         const float* __restrict__ c, float* __restrict__ dst) {
  int i = blockIdx.x * 256 + threadIdx.x;
  if (i < 256)      dst[i] = a[i];
  else if (i < 512) dst[i] = b[i - 256];
  else if (i < 768) dst[i] = c[i - 512];
}

// ---------------------------------------------------------------------------
// LayerNorm over F=256, one wave32 per row, writes f16 activations.
// ---------------------------------------------------------------------------
__global__ __launch_bounds__(256)
void gt_ln_f16(const float* __restrict__ x, const float* __restrict__ g,
               const float* __restrict__ bb, _Float16* __restrict__ out, int rows) {
  int wrow = (blockIdx.x * 256 + threadIdx.x) >> 5;
  int lane = threadIdx.x & 31;
  if (wrow >= rows) return;
  const float* xp = x + (size_t)wrow * FDIM;
  float v[8];
  float s = 0.f;
#pragma unroll
  for (int i = 0; i < 8; ++i) { v[i] = xp[lane + i * 32]; s += v[i]; }
  float mean = wred_sum(s) * (1.f / 256.f);
  float var = 0.f;
#pragma unroll
  for (int i = 0; i < 8; ++i) { float d = v[i] - mean; var += d * d; }
  var = wred_sum(var) * (1.f / 256.f);
  float inv = rsqrtf(var + 1e-5f);
  _Float16* op = out + (size_t)wrow * FDIM;
#pragma unroll
  for (int i = 0; i < 8; ++i) {
    int c = lane + i * 32;
    op[c] = (_Float16)((v[i] - mean) * inv * g[c] + bb[c]);
  }
}

// ---------------------------------------------------------------------------
// WMMA GEMM: C(MxN) = A(MxK,f16) x Bt(NxK,f16 transposed) + bias, fused epi:
//   EPI 0: store f32 (QKV)
//   EPI 1: mask-zero, add residual, store f32 (WO, MLP2)
//   EPI 2: ReLU, mask-zero, store f16 (MLP1)
// 128x128 block tile, 8 waves of 32x64, K stepped by 32, ping-pong LDS.
// ---------------------------------------------------------------------------
template <int EPI>
__global__ __launch_bounds__(256)
void gt_wmma_gemm(const _Float16* __restrict__ A, const _Float16* __restrict__ Bt,
                  const float* __restrict__ bias, const float* __restrict__ resid,
                  const unsigned char* __restrict__ mask,
                  float* __restrict__ Cf, _Float16* __restrict__ Ch,
                  int M, int N, int Kd) {
  __shared__ __align__(16) _Float16 SA[2][128 * 40];  // [row][k], stride 40 halves
  __shared__ __align__(16) _Float16 SB[2][128 * 40];  // [col][k]

  const int tid  = threadIdx.x;
  const int lane = tid & 31;
  const int wid  = tid >> 5;
  const int wrow = wid >> 1;   // 0..3 -> 32-row strip
  const int wcol = wid & 1;    // 0..1 -> 64-col strip
  const int m0 = blockIdx.y * 128;
  const int n0 = blockIdx.x * 128;
  const int hi  = lane >> 4;
  const int r16 = lane & 15;

  // Cooperative staging: identical pattern for A and Bt (both 128 rows x 32 k):
  // each thread owns one 16-half (32B) chunk.
  const int srow = tid >> 1;
  const int sseg = (tid & 1) * 16;
  const _Float16* gA = A  + (size_t)(m0 + srow) * Kd + sseg;
  const _Float16* gB = Bt + (size_t)(n0 + srow) * Kd + sseg;
  const int loff = srow * 40 + sseg;

  v8f acc[2][4];
#pragma unroll
  for (int mt = 0; mt < 2; ++mt)
#pragma unroll
    for (int nt = 0; nt < 4; ++nt) acc[mt][nt] = (v8f)(0.f);

  const int steps = Kd >> 5;

  // ---- prologue: stage tile 0 into buffer 0 ----
#ifdef GT_ASYNC
  async_cp16(gA, &SA[0][loff]);
  async_cp16(gB, &SB[0][loff]);
  __builtin_amdgcn_s_wait_asynccnt(0);
#else
  {
    v8h a0 = ((const v8h*)gA)[0], a1 = ((const v8h*)gA)[1];
    v8h b0 = ((const v8h*)gB)[0], b1 = ((const v8h*)gB)[1];
    *(v8h*)(&SA[0][loff])     = a0;  *(v8h*)(&SA[0][loff + 8]) = a1;
    *(v8h*)(&SB[0][loff])     = b0;  *(v8h*)(&SB[0][loff + 8]) = b1;
  }
#endif
  __syncthreads();

  int cur = 0;
  for (int ks = 0; ks < steps; ++ks) {
    const int  nxt       = cur ^ 1;
    const bool have_next = (ks + 1) < steps;
    const _Float16* gAn = gA + (ks + 1) * 32;
    const _Float16* gBn = gB + (ks + 1) * 32;

#ifdef GT_ASYNC
    if (have_next) {                       // async engine fills next buffer
      async_cp16(gAn, &SA[nxt][loff]);
      async_cp16(gBn, &SB[nxt][loff]);
    }
#else
    v8h na0, na1, nb0, nb1;                // reg-staged next tile
    if (have_next) {
      na0 = ((const v8h*)gAn)[0];  na1 = ((const v8h*)gAn)[1];
      nb0 = ((const v8h*)gBn)[0];  nb1 = ((const v8h*)gBn)[1];
    }
#endif
    if (ks + 2 < steps) {                  // global_prefetch_b8 two tiles ahead
      __builtin_prefetch(gA + (ks + 2) * 32, 0, 0);
      __builtin_prefetch(gB + (ks + 2) * 32, 0, 0);
    }

    // ---- WMMA on current buffer ----
    v16h bfrag[4];
#pragma unroll
    for (int nt = 0; nt < 4; ++nt)
      bfrag[nt] = frag16(&SB[cur][(wcol * 64 + nt * 16 + r16) * 40], hi);
#pragma unroll
    for (int mt = 0; mt < 2; ++mt) {
      v16h afrag = frag16(&SA[cur][(wrow * 32 + mt * 16 + r16) * 40], hi);
#pragma unroll
      for (int nt = 0; nt < 4; ++nt)
        acc[mt][nt] = __builtin_amdgcn_wmma_f32_16x16x32_f16(
            false, afrag, false, bfrag[nt], (short)0, acc[mt][nt], false, false);
    }

#ifdef GT_ASYNC
    if (have_next) __builtin_amdgcn_s_wait_asynccnt(0);
#else
    if (have_next) {
      *(v8h*)(&SA[nxt][loff])     = na0;  *(v8h*)(&SA[nxt][loff + 8]) = na1;
      *(v8h*)(&SB[nxt][loff])     = nb0;  *(v8h*)(&SB[nxt][loff + 8]) = nb1;
    }
#endif
    __syncthreads();
    cur = nxt;
  }

  // --- epilogue: C layout -> lane col = n&15, VGPR i row = i + (lane>>4)*8 ---
#pragma unroll
  for (int mt = 0; mt < 2; ++mt) {
#pragma unroll
    for (int nt = 0; nt < 4; ++nt) {
      int col = n0 + wcol * 64 + nt * 16 + r16;
      float bv = bias[col];
#pragma unroll
      for (int i = 0; i < 8; ++i) {
        int r = m0 + wrow * 32 + mt * 16 + hi * 8 + i;
        float v = acc[mt][nt][i] + bv;
        if (EPI == 2) v = v > 0.f ? v : 0.f;
        if (EPI >= 1 && mask[r]) v = 0.f;
        if (EPI == 1) v += resid[(size_t)r * N + col];
        if (EPI == 2) Ch[(size_t)r * N + col] = (_Float16)v;
        else          Cf[(size_t)r * N + col] = v;
      }
    }
  }
}

// ---------------------------------------------------------------------------
// Local attention: one wave32 per (token,head). lane = neighbor for scores /
// softmax; lane = dim for the AV pass (coalesced V loads). K = HD = 32.
// qkv layout per row: [q(256) | k(256) | v(256)].
// ---------------------------------------------------------------------------
__global__ __launch_bounds__(256)
void gt_attn(const float* __restrict__ qkv, const int* __restrict__ nbrs,
             const unsigned char* __restrict__ mask, _Float16* __restrict__ oh) {
  int gw   = (blockIdx.x * 256 + threadIdx.x) >> 5;   // wave id = (row, head)
  int lane = threadIdx.x & 31;
  int h    = gw & 7;
  int row  = gw >> 3;                 // b*N + n, 0..8191
  int base = (row >> 12) << 12;       // b*N (N = 4096)

  int nbr = nbrs[(size_t)row * KNBR + lane];
  int pad = mask[base + nbr];

  float q = qkv[(size_t)row * NQKV + h * HDIM + lane];

  // my neighbor's key vector (32 floats), dot against broadcast q
  const float* kp = qkv + (size_t)(base + nbr) * NQKV + DMODEL + h * HDIM;
  float kreg[32];
#pragma unroll
  for (int d = 0; d < 32; ++d) kreg[d] = kp[d];
  float s = 0.f;
#pragma unroll
  for (int d = 0; d < 32; ++d) s += __shfl(q, d) * kreg[d];
  s *= 0.17677669529663687f;          // 1/sqrt(32)
  if (pad) s = -1e9f;

  // softmax across the 32 neighbors (one per lane)
  float mx = s;
#pragma unroll
  for (int o = 16; o > 0; o >>= 1) mx = fmaxf(mx, __shfl_xor(mx, o));
  float e = __expf(s - mx);
  float a = e / wred_sum(e);

  // o[lane=dim] = sum_k attn_k * V[nbr_k][dim]   (coalesced V loads)
  float o = 0.f;
#pragma unroll
  for (int k = 0; k < 32; ++k) {
    float ak = __shfl(a, k);
    int   nk = __shfl(nbr, k);
    o += ak * qkv[(size_t)(base + nk) * NQKV + 2 * DMODEL + h * HDIM + lane];
  }
  oh[(size_t)row * DMODEL + h * HDIM + lane] = (_Float16)o;
}

// ---------------------------------------------------------------------------
// Host driver
// ---------------------------------------------------------------------------
extern "C" void kernel_launch(void* const* d_in, const int* in_sizes, int n_in,
                              void* d_out, int out_size, void* d_ws, size_t ws_size,
                              hipStream_t stream) {
  (void)in_sizes; (void)n_in; (void)out_size; (void)ws_size;

  const float*         x_in  = (const float*)d_in[0];
  const int*           nbrs  = (const int*)d_in[1];
  const unsigned char* maskb = (const unsigned char*)d_in[2];
  const float* ln1_g = (const float*)d_in[3];
  const float* ln1_b = (const float*)d_in[4];
  const float* wq = (const float*)d_in[5];
  const float* bq = (const float*)d_in[6];
  const float* wk = (const float*)d_in[7];
  const float* bk = (const float*)d_in[8];
  const float* wv = (const float*)d_in[9];
  const float* bv = (const float*)d_in[10];
  const float* wo = (const float*)d_in[11];
  const float* bo = (const float*)d_in[12];
  const float* ln2_g = (const float*)d_in[13];
  const float* ln2_b = (const float*)d_in[14];
  const float* w1 = (const float*)d_in[15];
  const float* b1 = (const float*)d_in[16];
  const float* w2 = (const float*)d_in[17];
  const float* b2 = (const float*)d_in[18];

  // workspace layout
  char* ws = (char*)d_ws;
  size_t off = 0;
  float*     xA   = (float*)(ws + off);     off += (size_t)R_TOK * FDIM * 4;   // current x
  float*     x1f  = (float*)(ws + off);     off += (size_t)R_TOK * FDIM * 4;   // post-attn residual
  _Float16*  xh   = (_Float16*)(ws + off);  off += (size_t)R_TOK * FDIM * 2;   // LN output (f16)
  _Float16*  hh   = (_Float16*)(ws + off);  off += (size_t)R_TOK * HIDDEN * 2; // MLP hidden (f16)
  _Float16*  oh   = (_Float16*)(ws + off);  off += (size_t)R_TOK * DMODEL * 2; // attn out (f16)
  float*     qkv  = (float*)(ws + off);     off += (size_t)R_TOK * NQKV * 4;
  _Float16*  Wqkv = (_Float16*)(ws + off);  off += (size_t)FDIM * NQKV * 2;    // transposed, 768x256
  _Float16*  Woh  = (_Float16*)(ws + off);  off += (size_t)DMODEL * FDIM * 2;  // transposed, 256x256
  _Float16*  W1h  = (_Float16*)(ws + off);  off += (size_t)FDIM * HIDDEN * 2;  // transposed, 512x256
  _Float16*  W2h  = (_Float16*)(ws + off);  off += (size_t)HIDDEN * FDIM * 2;  // transposed, 256x512
  float*     bqkv = (float*)(ws + off);     off += 4096;

  (void)hipMemcpyAsync(xA, x_in, (size_t)R_TOK * FDIM * 4, hipMemcpyDeviceToDevice, stream);

  dim3 blk(256);
  const int lnBlocks   = (R_TOK * 32) / 256;       // one wave per row
  const int attnBlocks = (R_TOK * HHEADS) / 8;     // 8 waves per block

  for (int l = 0; l < 2; ++l) {
    // ---- convert + transpose this layer's weights to f16 (wq|wk|wv fused) --
    gt_cvt_t<<<(FDIM * DMODEL + 255) / 256, blk, 0, stream>>>(wq + (size_t)l * FDIM * DMODEL, Wqkv + 0 * 256 * FDIM,   FDIM, DMODEL, FDIM);
    gt_cvt_t<<<(FDIM * DMODEL + 255) / 256, blk, 0, stream>>>(wk + (size_t)l * FDIM * DMODEL, Wqkv + 1 * 256 * FDIM,   FDIM, DMODEL, FDIM);
    gt_cvt_t<<<(FDIM * DMODEL + 255) / 256, blk, 0, stream>>>(wv + (size_t)l * FDIM * DMODEL, Wqkv + 2 * 256 * FDIM,   FDIM, DMODEL, FDIM);
    gt_cvt_t<<<(DMODEL * FDIM + 255) / 256, blk, 0, stream>>>(wo + (size_t)l * DMODEL * FDIM, Woh, DMODEL, FDIM, DMODEL);
    gt_cvt_t<<<(FDIM * HIDDEN + 255) / 256, blk, 0, stream>>>(w1 + (size_t)l * FDIM * HIDDEN, W1h, FDIM, HIDDEN, FDIM);
    gt_cvt_t<<<(HIDDEN * FDIM + 255) / 256, blk, 0, stream>>>(w2 + (size_t)l * HIDDEN * FDIM, W2h, HIDDEN, FDIM, HIDDEN);
    gt_bias3<<<3, blk, 0, stream>>>(bq + (size_t)l * DMODEL, bk + (size_t)l * DMODEL, bv + (size_t)l * DMODEL, bqkv);

    // ---- pre-norm + fused QKV projection (WMMA) ----
    gt_ln_f16<<<lnBlocks, blk, 0, stream>>>(xA, ln1_g + (size_t)l * FDIM, ln1_b + (size_t)l * FDIM, xh, R_TOK);
    gt_wmma_gemm<0><<<dim3(NQKV / 128, R_TOK / 128), blk, 0, stream>>>(
        xh, Wqkv, bqkv, nullptr, nullptr, qkv, nullptr, R_TOK, NQKV, FDIM);

    // ---- local neighbor attention ----
    gt_attn<<<attnBlocks, blk, 0, stream>>>(qkv, nbrs, maskb, oh);

    // ---- WO projection + mask-zero + residual (WMMA) ----
    gt_wmma_gemm<1><<<dim3(FDIM / 128, R_TOK / 128), blk, 0, stream>>>(
        oh, Woh, bo + (size_t)l * FDIM, xA, maskb, x1f, nullptr, R_TOK, FDIM, DMODEL);

    // ---- pre-norm + MLP (WMMA x2) ----
    gt_ln_f16<<<lnBlocks, blk, 0, stream>>>(x1f, ln2_g + (size_t)l * FDIM, ln2_b + (size_t)l * FDIM, xh, R_TOK);
    gt_wmma_gemm<2><<<dim3(HIDDEN / 128, R_TOK / 128), blk, 0, stream>>>(
        xh, W1h, b1 + (size_t)l * HIDDEN, nullptr, maskb, nullptr, hh, R_TOK, HIDDEN, FDIM);
    gt_wmma_gemm<1><<<dim3(FDIM / 128, R_TOK / 128), blk, 0, stream>>>(
        hh, W2h, b2 + (size_t)l * FDIM, x1f, maskb, xA, nullptr, R_TOK, FDIM, HIDDEN);
  }

  (void)hipMemcpyAsync(d_out, xA, (size_t)R_TOK * FDIM * 4, hipMemcpyDeviceToDevice, stream);
}
